// ResidualMultiHeadAttention_52252572123762
// MI455X (gfx1250) — matrix operand order repaired and checked by reading, not compiled, not run on hardware
//
#include <hip/hip_runtime.h>
#include <hip/hip_bf16.h>

// ---------------- problem constants ----------------
#define BB      2
#define SS      2048
#define DD      2048
#define HH      16
#define KVH     4
#define DEPTH   128
#define KVD     512          // KV_DIM
#define GROUPS  4
#define MROWS   (BB * SS)    // 4096
#define SCALE   0.08838834764831845f   // 1/sqrt(128)
#define MASKF   (-10000.0f)
#define LN_EPS  1e-6f

// CDNA5 async global->LDS staging of the V tile (double buffered).
#define USE_ASYNC_V 1

// ---------------- WMMA plumbing ----------------
typedef __attribute__((ext_vector_type(16))) __bf16 bf16x16;
typedef __attribute__((ext_vector_type(8)))  float  v8f;

union FragBF {
    bf16x16 v;
    uint4   q[2];
};

static __device__ __forceinline__ v8f wmma_bf16(const FragBF& a, const FragBF& b, v8f c) {
    return __builtin_amdgcn_wmma_f32_16x16x32_bf16(
        false, a.v, false, b.v, (short)0, c, false, false);
}

// A-operand fragment: per-lane 16 contiguous bf16 (K {0..15}/{16..31} by lane half)
static __device__ __forceinline__ void ldfrag_a(FragBF& f, const unsigned short* p) {
    f.q[0] = *reinterpret_cast<const uint4*>(p);
    f.q[1] = *reinterpret_cast<const uint4*>(p + 8);
}
// B-operand fragment: per-lane 16 contiguous bf16 (K {0..15}/{16..31} by lane half)
static __device__ __forceinline__ void ldfrag_b(FragBF& f, const unsigned short* p) {
    f.q[0] = *reinterpret_cast<const uint4*>(p);
    f.q[1] = *reinterpret_cast<const uint4*>(p + 8);
}

static __device__ __forceinline__ unsigned short f2bf(float f) {
    unsigned int u = __float_as_uint(f);
    u += 0x7FFFu + ((u >> 16) & 1u);     // round-to-nearest-even
    return (unsigned short)(u >> 16);
}

// ---------------- elementwise f32 -> bf16 ----------------
__global__ void cvt_f32_bf16(const float* __restrict__ in,
                             unsigned short* __restrict__ out, int n4) {
    int i = blockIdx.x * blockDim.x + threadIdx.x;
    if (i >= n4) return;
    float4 f = reinterpret_cast<const float4*>(in)[i];
    ushort4 o;
    o.x = f2bf(f.x); o.y = f2bf(f.y); o.z = f2bf(f.z); o.w = f2bf(f.w);
    reinterpret_cast<ushort4*>(out)[i] = o;
}

// ---------------- transpose+convert: W[K][N] f32 -> Wt[N][K] bf16 ----------------
__global__ void transpose_w(const float* __restrict__ in,
                            unsigned short* __restrict__ out, int Kd, int Nn) {
    __shared__ float tile[32][33];
    int tx = threadIdx.x, ty = threadIdx.y;
    int n  = blockIdx.x * 32 + tx;
    int k0 = blockIdx.y * 32;
#pragma unroll
    for (int j = ty; j < 32; j += 8)
        tile[j][tx] = in[(size_t)(k0 + j) * Nn + n];
    __syncthreads();
    int kout = k0 + tx;
    int nb   = blockIdx.x * 32;
#pragma unroll
    for (int j = ty; j < 32; j += 8)
        out[(size_t)(nb + j) * Kd + kout] = f2bf(tile[tx][j]);
}

// ---------------- V transpose (bf16): Vb[M][KVD] -> Vt[b][kvh][DEPTH][S] ----------------
__global__ void transpose_v(const unsigned short* __restrict__ Vb,
                            unsigned short* __restrict__ Vt) {
    __shared__ unsigned short tile[32][33];
    int tx = threadIdx.x, ty = threadIdx.y;
    int z  = blockIdx.z;
    int b  = z / KVH, kvh = z % KVH;
    int s0 = blockIdx.x * 32;
    int d0 = blockIdx.y * 32;
#pragma unroll
    for (int j = ty; j < 32; j += 8)
        tile[j][tx] = Vb[(size_t)(b * SS + s0 + j) * KVD + kvh * DEPTH + d0 + tx];
    __syncthreads();
    unsigned short* plane = Vt + (size_t)(b * KVH + kvh) * DEPTH * SS;
#pragma unroll
    for (int j = ty; j < 32; j += 8)
        plane[(size_t)(d0 + j) * SS + s0 + tx] = tile[tx][j];
}

// ---------------- WMMA GEMM: C[M][N] = A[M][K] * Bt[N][K]^T + bias ----------------
// Wave tile 64(M) x 32(N); fragments double-buffered across k-chunks so the
// 12 global_load_b128 of chunk k+1 overlap the 8 WMMAs of chunk k.
// MODE 0: bf16 out.   MODE 1: f32 out + residual add.
template <int MODE>
__global__ void gemm_wmma(const unsigned short* __restrict__ A,
                          const unsigned short* __restrict__ Bt,
                          const float* __restrict__ bias,
                          const float* __restrict__ resid,
                          void* __restrict__ out, int Nn, int Kd) {
    const int lane = threadIdx.x;
    const int half = lane >> 4;
    const int l15  = lane & 15;
    const int n0 = blockIdx.x * 32;
    const int m0 = blockIdx.y * 64;

    v8f acc[4][2];
#pragma unroll
    for (int s = 0; s < 4; ++s) {
        acc[s][0] = (v8f){0,0,0,0,0,0,0,0};
        acc[s][1] = (v8f){0,0,0,0,0,0,0,0};
    }

    const unsigned short* ar[4];
#pragma unroll
    for (int s = 0; s < 4; ++s)
        ar[s] = A + (size_t)(m0 + 16 * s + l15) * Kd + half * 16;
    const unsigned short* br0 = Bt + (size_t)(n0 + l15) * Kd + half * 16;
    const unsigned short* br1 = br0 + (size_t)16 * Kd;

    FragBF a[2][4], bb[2][2];
    // prime chunk 0 into buffer 0
    ldfrag_b(bb[0][0], br0);
    ldfrag_b(bb[0][1], br1);
#pragma unroll
    for (int s = 0; s < 4; ++s) ldfrag_a(a[0][s], ar[s]);

    int idx = 0;
    for (int k0 = 0; k0 < Kd; k0 += 32, idx ^= 1) {
        if (k0 + 32 < Kd) {
            int nk = k0 + 32, nx = idx ^ 1;
            ldfrag_b(bb[nx][0], br0 + nk);
            ldfrag_b(bb[nx][1], br1 + nk);
#pragma unroll
            for (int s = 0; s < 4; ++s) ldfrag_a(a[nx][s], ar[s] + nk);
        }
#pragma unroll
        for (int s = 0; s < 4; ++s) {
            acc[s][0] = wmma_bf16(a[idx][s], bb[idx][0], acc[s][0]);
            acc[s][1] = wmma_bf16(a[idx][s], bb[idx][1], acc[s][1]);
        }
    }

    const int col0 = n0 + l15, col1 = col0 + 16;
    const float bias0 = bias[col0], bias1 = bias[col1];
#pragma unroll
    for (int s = 0; s < 4; ++s) {
#pragma unroll
        for (int i = 0; i < 8; ++i) {
            int row = m0 + 16 * s + i + 8 * half;
            float v0 = acc[s][0][i] + bias0;
            float v1 = acc[s][1][i] + bias1;
            if (MODE == 0) {
                unsigned short* O = (unsigned short*)out;
                O[(size_t)row * Nn + col0] = f2bf(v0);
                O[(size_t)row * Nn + col1] = f2bf(v1);
            } else {
                float* O = (float*)out;
                O[(size_t)row * Nn + col0] = v0 + resid[(size_t)row * Nn + col0];
                O[(size_t)row * Nn + col1] = v1 + resid[(size_t)row * Nn + col1];
            }
        }
    }
}

// ---------------- flash attention (causal GQA), one wave per 16-query block --------
// Qb [M][D] bf16, Kb [M][KVD] bf16, Vt [B][KVH][DEPTH][S] bf16, Ab [M][D] bf16
__global__ void flash_attn(const unsigned short* __restrict__ Qb,
                           const unsigned short* __restrict__ Kb,
                           const unsigned short* __restrict__ Vt,
                           unsigned short* __restrict__ Ab) {
    __shared__ float Slds[16 * 33];                       // 16x32 logits (pad 33)
    __shared__ __align__(16) float Corr[16];
    __shared__ __align__(16) float Lrow[16];
    __shared__ unsigned short Plds[16 * 40];              // 16x32 bf16 probs (pad 40)
#if USE_ASYNC_V
    __shared__ __align__(16) unsigned short Vbuf[2][DEPTH * 32];  // 2 x 8KB V tiles
#endif

    const int lane = threadIdx.x;
    const int half = lane >> 4;
    const int r    = lane & 15;

    const int bh  = blockIdx.y;
    const int b   = bh / HH, h = bh % HH;
    const int kvh = h / GROUPS;
    const int q0  = blockIdx.x * 16;
    const int qidx = q0 + r;

    // ---- Q A-fragments (16 rows x 128 depth = 4 frags), register-resident ----
    const unsigned short* qptr = Qb + (size_t)(b * SS + q0 + r) * DD + h * DEPTH + half * 16;
    FragBF qa[4];
#pragma unroll
    for (int t = 0; t < 4; ++t) ldfrag_a(qa[t], qptr + t * 32);

    v8f o[8];
#pragma unroll
    for (int t = 0; t < 8; ++t) o[t] = (v8f){0,0,0,0,0,0,0,0};
    float m_i = -1e30f, l_i = 0.0f;

    const unsigned short* kbase  = Kb + (size_t)(b * SS) * KVD + kvh * DEPTH;
    const unsigned short* vplane = Vt + (size_t)(b * KVH + kvh) * DEPTH * SS;

#if USE_ASYNC_V
    // per-lane async mapping: 16 b128 ops move one 128x32 bf16 tile (8KB)
    const int arow = lane >> 2;            // 0..7 row within 8-row group
    const int abyt = (lane & 3) * 16;      // byte offset within 64B row
    const unsigned vbase = (unsigned)(uintptr_t)(&Vbuf[0][0]);
#endif

    const int jbmax = (q0 + 15) / 32;      // inclusive; 32-key tiles
#if USE_ASYNC_V
    {   // prologue: stage V tile for jb = 0 into buffer 0
        unsigned long long g0 = (unsigned long long)(uintptr_t)vplane + (size_t)arow * (SS * 2) + abyt;
#pragma unroll
        for (int i = 0; i < 16; ++i) {
            unsigned long long ga = g0 + (size_t)(i * 8) * (SS * 2);
            unsigned la = vbase + (unsigned)((i * 8 + arow) * 64 + abyt);
            asm volatile("global_load_async_to_lds_b128 %0, %1, off"
                         :: "v"(la), "v"(ga) : "memory");
        }
    }
#endif

    for (int jb = 0; jb <= jbmax; ++jb) {
        const int kb0 = jb * 32;
#if USE_ASYNC_V
        if (jb < jbmax) {  // stage next V tile into the other buffer
            unsigned long long g0 = (unsigned long long)(uintptr_t)vplane
                                  + (size_t)arow * (SS * 2) + (size_t)(kb0 + 32) * 2 + abyt;
            unsigned lb = vbase + (unsigned)(((jb + 1) & 1) * (DEPTH * 32 * 2));
#pragma unroll
            for (int i = 0; i < 16; ++i) {
                unsigned long long ga = g0 + (size_t)(i * 8) * (SS * 2);
                unsigned la = lb + (unsigned)((i * 8 + arow) * 64 + abyt);
                asm volatile("global_load_async_to_lds_b128 %0, %1, off"
                             :: "v"(la), "v"(ga) : "memory");
            }
        }
#endif
        if (jb < jbmax)    // gfx1250 prefetch of next K tile
            __builtin_prefetch(kbase + (size_t)(kb0 + 32 + r) * KVD, 0, 1);

        // ---- logits: S = Q Kt (two 16x16 tiles; K-dim 128 in 4 chunks, pipelined) ----
        v8f c0 = (v8f){0,0,0,0,0,0,0,0}, c1 = c0;
        const unsigned short* kr0 = kbase + (size_t)(kb0 + r) * KVD + half * 16;
        const unsigned short* kr1 = kr0 + (size_t)16 * KVD;
        FragBF kf0[2], kf1[2];
        ldfrag_b(kf0[0], kr0);
        ldfrag_b(kf1[0], kr1);
#pragma unroll
        for (int t = 0; t < 4; ++t) {
            int cur = t & 1;
            if (t < 3) {
                ldfrag_b(kf0[cur ^ 1], kr0 + (t + 1) * 32);
                ldfrag_b(kf1[cur ^ 1], kr1 + (t + 1) * 32);
            }
            c0 = wmma_bf16(qa[t], kf0[cur], c0);
            c1 = wmma_bf16(qa[t], kf1[cur], c1);
        }

        // ---- spill logits (C layout) to LDS row-major 16x32 ----
#pragma unroll
        for (int i = 0; i < 8; ++i) {
            int row = i + 8 * half;
            Slds[row * 33 + r]      = c0[i];
            Slds[row * 33 + 16 + r] = c1[i];
        }
        __syncthreads();

        // ---- online softmax: lane pair (r, r+16) owns row r, 16 cols each ----
        // pass 1: row max (re-read LDS, keep register pressure low)
        float tmax = -1e30f;
#pragma unroll
        for (int cc = 0; cc < 16; ++cc) {
            int key = kb0 + half * 16 + cc;
            float v = Slds[r * 33 + half * 16 + cc] * SCALE;
            if (key > qidx) v = MASKF;
            tmax = fmaxf(tmax, v);
        }
        tmax = fmaxf(tmax, __shfl_xor(tmax, 16, 32));
        float m_new = fmaxf(m_i, tmax);
        float corr  = __expf(m_i - m_new);
        // pass 2: exp, pack to bf16, store immediately
        float psum = 0.0f;
        unsigned int* prow = reinterpret_cast<unsigned int*>(Plds + r * 40 + half * 16);
#pragma unroll
        for (int cc = 0; cc < 8; ++cc) {
            int key0 = kb0 + half * 16 + 2 * cc;
            float v0 = Slds[r * 33 + half * 16 + 2 * cc]     * SCALE;
            float v1 = Slds[r * 33 + half * 16 + 2 * cc + 1] * SCALE;
            if (key0     > qidx) v0 = MASKF;
            if (key0 + 1 > qidx) v1 = MASKF;
            float p0 = __expf(v0 - m_new);
            float p1 = __expf(v1 - m_new);
            psum += p0 + p1;
            prow[cc] = (unsigned int)f2bf(p0) | ((unsigned int)f2bf(p1) << 16);
        }
        psum += __shfl_xor(psum, 16, 32);
        l_i = l_i * corr + psum;
        m_i = m_new;
        if (half == 0) Corr[r] = corr;
        __syncthreads();

        // ---- rescale output accumulators by per-row correction ----
        float4 ca = *reinterpret_cast<const float4*>(&Corr[8 * half]);
        float4 cb = *reinterpret_cast<const float4*>(&Corr[8 * half + 4]);
        float cr[8] = {ca.x, ca.y, ca.z, ca.w, cb.x, cb.y, cb.z, cb.w};
#pragma unroll
        for (int t = 0; t < 8; ++t)
#pragma unroll
            for (int i = 0; i < 8; ++i) o[t][i] *= cr[i];

        // ---- P fragment (A layout) from LDS, then O += P * V ----
        FragBF pf;
        ldfrag_a(pf, Plds + r * 40 + half * 16);

#if USE_ASYNC_V
        if (jb < jbmax) asm volatile("s_wait_asynccnt 16" ::: "memory");
        else            asm volatile("s_wait_asynccnt 0"  ::: "memory");
        const unsigned short* vtile = &Vbuf[jb & 1][0];
        FragBF vf[2];
        ldfrag_b(vf[0], vtile + r * 32 + half * 16);
#pragma unroll
        for (int t = 0; t < 8; ++t) {
            int cur = t & 1;
            if (t < 7)
                ldfrag_b(vf[cur ^ 1], vtile + ((t + 1) * 16 + r) * 32 + half * 16);
            o[t] = wmma_bf16(pf, vf[cur], o[t]);
        }
#else
        const unsigned short* vr0 = vplane + (size_t)r * SS + kb0 + half * 16;
        FragBF vf[2];
        ldfrag_b(vf[0], vr0);
#pragma unroll
        for (int t = 0; t < 8; ++t) {
            int cur = t & 1;
            if (t < 7) ldfrag_b(vf[cur ^ 1], vr0 + (size_t)((t + 1) * 16) * SS);
            o[t] = wmma_bf16(pf, vf[cur], o[t]);
        }
#endif
    }

    // ---- finalize: divide by l, store merged-heads bf16 ----
    if (half == 0) Lrow[r] = l_i;
    __syncthreads();
    float4 la = *reinterpret_cast<const float4*>(&Lrow[8 * half]);
    float4 lb = *reinterpret_cast<const float4*>(&Lrow[8 * half + 4]);
    float inv[8] = {1.0f/la.x, 1.0f/la.y, 1.0f/la.z, 1.0f/la.w,
                    1.0f/lb.x, 1.0f/lb.y, 1.0f/lb.z, 1.0f/lb.w};
    unsigned short* obase = Ab + (size_t)(b * SS + q0) * DD + h * DEPTH;
#pragma unroll
    for (int t = 0; t < 8; ++t)
#pragma unroll
        for (int i = 0; i < 8; ++i) {
            int row = i + 8 * half;
            obase[(size_t)row * DD + t * 16 + r] = f2bf(o[t][i] * inv[i]);
        }
}

// ---------------- LayerNorm over D=2048 per row ----------------
__global__ void layernorm_k(const float* __restrict__ X,
                            const float* __restrict__ gamma,
                            const float* __restrict__ beta,
                            float* __restrict__ out) {
    __shared__ float s1[256], s2[256];
    const int row = blockIdx.x;
    const int tid = threadIdx.x;
    const float* xr = X + (size_t)row * DD;
    float sum = 0.0f, sq = 0.0f;
#pragma unroll
    for (int j = 0; j < DD / 256; ++j) {
        float v = xr[tid + j * 256];
        sum += v; sq += v * v;
    }
    s1[tid] = sum; s2[tid] = sq;
    __syncthreads();
    for (int off = 128; off > 0; off >>= 1) {
        if (tid < off) { s1[tid] += s1[tid + off]; s2[tid] += s2[tid + off]; }
        __syncthreads();
    }
    float mean = s1[0] * (1.0f / DD);
    float var  = s2[0] * (1.0f / DD) - mean * mean;
    float rstd = rsqrtf(var + LN_EPS);
    float* orow = out + (size_t)row * DD;
#pragma unroll
    for (int j = 0; j < DD / 256; ++j) {
        int c = tid + j * 256;
        orow[c] = (xr[c] - mean) * rstd * gamma[c] + beta[c];
    }
}

// ---------------- workspace layout (bytes) ----------------
#define OFF_XB   ((size_t)0)
#define OFF_WQT  ((size_t)16777216)
#define OFF_WKT  ((size_t)25165824)
#define OFF_WVT  ((size_t)27262976)
#define OFF_WOT  ((size_t)29360128)
#define OFF_QB   ((size_t)37748736)
#define OFF_KB   ((size_t)54525952)
#define OFF_VB   ((size_t)58720256)
#define OFF_VT   ((size_t)62914560)
#define OFF_AB   ((size_t)67108864)
#define OFF_XR   ((size_t)83886080)
#define WS_NEED  ((size_t)117440512)

extern "C" void kernel_launch(void* const* d_in, const int* in_sizes, int n_in,
                              void* d_out, int out_size, void* d_ws, size_t ws_size,
                              hipStream_t stream) {
    if (ws_size < WS_NEED) return;
    const float* query = (const float*)d_in[0];
    // d_in[1] = mask (causal, applied analytically)
    const float* wq = (const float*)d_in[2];
    const float* bq = (const float*)d_in[3];
    const float* wk = (const float*)d_in[4];
    const float* bk = (const float*)d_in[5];
    const float* wv = (const float*)d_in[6];
    const float* bv = (const float*)d_in[7];
    const float* wo = (const float*)d_in[8];
    const float* bo = (const float*)d_in[9];
    const float* gamma = (const float*)d_in[10];
    const float* beta  = (const float*)d_in[11];
    float* out = (float*)d_out;

    char* ws = (char*)d_ws;
    unsigned short* Xb  = (unsigned short*)(ws + OFF_XB);
    unsigned short* WqT = (unsigned short*)(ws + OFF_WQT);
    unsigned short* WkT = (unsigned short*)(ws + OFF_WKT);
    unsigned short* WvT = (unsigned short*)(ws + OFF_WVT);
    unsigned short* WoT = (unsigned short*)(ws + OFF_WOT);
    unsigned short* Qb  = (unsigned short*)(ws + OFF_QB);
    unsigned short* Kb  = (unsigned short*)(ws + OFF_KB);
    unsigned short* Vb  = (unsigned short*)(ws + OFF_VB);
    unsigned short* Vt  = (unsigned short*)(ws + OFF_VT);
    unsigned short* Ab  = (unsigned short*)(ws + OFF_AB);
    float*          Xr  = (float*)(ws + OFF_XR);

    {   // 1. query f32 -> bf16
        int n4 = MROWS * DD / 4;
        cvt_f32_bf16<<<(n4 + 255) / 256, 256, 0, stream>>>(query, Xb, n4);
    }
    {   // 2. weight transposes (f32 [K][N] -> bf16 [N][K])
        dim3 blk(32, 8);
        transpose_w<<<dim3(DD / 32,  DD / 32), blk, 0, stream>>>(wq, WqT, DD, DD);
        transpose_w<<<dim3(KVD / 32, DD / 32), blk, 0, stream>>>(wk, WkT, DD, KVD);
        transpose_w<<<dim3(KVD / 32, DD / 32), blk, 0, stream>>>(wv, WvT, DD, KVD);
        transpose_w<<<dim3(DD / 32,  DD / 32), blk, 0, stream>>>(wo, WoT, DD, DD);
    }
    // 3. projections (WMMA bf16), wave tile 64x32, k-pipelined
    gemm_wmma<0><<<dim3(DD / 32,  MROWS / 64), 32, 0, stream>>>(Xb, WqT, bq, nullptr, Qb, DD,  DD);
    gemm_wmma<0><<<dim3(KVD / 32, MROWS / 64), 32, 0, stream>>>(Xb, WkT, bk, nullptr, Kb, KVD, DD);
    gemm_wmma<0><<<dim3(KVD / 32, MROWS / 64), 32, 0, stream>>>(Xb, WvT, bv, nullptr, Vb, KVD, DD);
    // 4. V -> Vt [b][kvh][d][s]
    transpose_v<<<dim3(SS / 32, DEPTH / 32, BB * KVH), dim3(32, 8), 0, stream>>>(Vb, Vt);
    // 5. causal GQA flash attention (async V staging)
    flash_attn<<<dim3(SS / 16, BB * HH), 32, 0, stream>>>(Qb, Kb, Vt, Ab);
    // 6. output projection + bias + residual (f32 out)
    gemm_wmma<1><<<dim3(DD / 32, MROWS / 64), 32, 0, stream>>>(Ab, WoT, bo, query, Xr, DD, DD);
    // 7. layernorm -> d_out
    layernorm_k<<<MROWS, 256, 0, stream>>>(Xr, gamma, beta, out);
}
